// FeedBack_Forecasting_75187697483909
// MI455X (gfx1250) — matrix operand order repaired and checked by reading, not compiled, not run on hardware
//
#include <hip/hip_runtime.h>
#include <math.h>

// ---------------- constants from the reference ----------------
#define Bsz   8192
#define Tt    70
#define Ff    64
#define Hh    128
#define LAGc  14
#define GAMMAc 28
#define WARM  (Tt - LAGc)      // 56
#define KDIM  (Ff + Hh)        // 192 : fused [x|h] K dimension
#define NZ    (4 * Hh)         // 512 : 4 gates
#define NWc   (Hh * Hh + Hh)   // 16512

// ---------------- WMMA vector types (gfx1250, wave32) ----------------
typedef __attribute__((ext_vector_type(16))) __bf16 v16bf;
typedef __attribute__((ext_vector_type(8)))  __bf16 v8bf;
typedef __attribute__((ext_vector_type(8)))  float  v8f;

union AFrag { v16bf v; v8bf h[2]; };

// ---------------- workspace layout (bytes) ----------------
#define WS_OFF_KT    196608u     // Kt bf16 [28][128][128]
#define WS_OFF_BVEC  1114112u    // bvec f32 [28][128]

__device__ __forceinline__ float sig_(float x) { return 1.0f / (1.0f + __expf(-x)); }
__device__ __forceinline__ float softplus_(float x) { return log1pf(__expf(x)); }

// ============ CDNA5 async global->LDS helpers (ASYNCcnt tracked) ============
__device__ __forceinline__ void async_copy16(unsigned ldsDst, const void* gsrc)
{
    // volatile pins the instruction; ordering vs LDS readers is provided by
    // s_wait_asynccnt + __syncthreads() at the consumption points.
    asm volatile("global_load_async_to_lds_b128 %0, %1, off"
                 :: "v"(ldsDst), "v"((unsigned long long)(uintptr_t)gsrc));
}
__device__ __forceinline__ void wait_async0()
{
    asm volatile("s_wait_asynccnt 0x0" ::: "memory");
}
__device__ __forceinline__ unsigned lds_addr(const void* p)
{
    return (unsigned)(uintptr_t)p;   // low 32 bits of LDS aperture address
}

// 16 rows x 64 f32 input-row tile (rows strided by T*F in global)
__device__ __forceinline__ void prefetch_input_row(const float* __restrict__ inputs,
                                                   int b0, int t, float* sDst, int tid)
{
    const int r = tid >> 4, c = tid & 15;
    const float* g = inputs + ((size_t)(b0 + r) * Tt + t) * Ff + c * 4;
    async_copy16(lds_addr(sDst) + tid * 16, g);
}
// contiguous 4KB tile (eps_y[s] block rows are contiguous in batch)
__device__ __forceinline__ void prefetch_flat4k(const float* __restrict__ gbase,
                                                float* sDst, int tid)
{
    async_copy16(lds_addr(sDst) + tid * 16, gbase + tid * 4);
}
// 32KB Kt step matrix: 2048 x 16B chunks, 8 per thread
__device__ __forceinline__ void prefetch_kt(const __bf16* __restrict__ ktStep,
                                            __bf16* sDst, int tid)
{
    const unsigned base = lds_addr(sDst);
#pragma unroll
    for (int c = 0; c < 8; ++c) {
        const int i = c * 256 + tid;
        async_copy16(base + i * 16, (const char*)ktStep + (size_t)i * 16);
    }
}

// ============ precompute: transpose weights + sample dense layers ============
__global__ void __launch_bounds__(256)
precompute_kernel(const float* __restrict__ lstm_W,   // [64][512]
                  const float* __restrict__ lstm_U,   // [128][512]
                  const float* __restrict__ post_t,   // [2*NW]
                  const float* __restrict__ eps_w,    // [28][NW]
                  __bf16* __restrict__ WtUt,          // [512][192]  Bt layout
                  __bf16* __restrict__ Kt,            // [28][128][128] Bt layout
                  float*  __restrict__ bvec)          // [28][128]
{
    const int idx = blockIdx.x * blockDim.x + threadIdx.x;

    if (idx < NZ * KDIM) {     // WtUt[n][k] = (k<64 ? W[k][n] : U[k-64][n])
        const int n = idx / KDIM, k = idx % KDIM;
        const float v = (k < Ff) ? lstm_W[k * NZ + n] : lstm_U[(k - Ff) * NZ + n];
        WtUt[idx] = (__bf16)v;
    }
    if (idx < GAMMAc * Hh * Hh) {   // Kt[t][n][k] = w_t[k*128 + n]
        const int t   = idx / (Hh * Hh);
        const int rem = idx % (Hh * Hh);
        const int n   = rem / Hh;
        const int k   = rem % Hh;
        const int widx = k * Hh + n;
        const float loc   = post_t[widx];
        const float rho   = post_t[NWc + widx];
        const float scale = 1e-5f + 0.1f * softplus_(rho);
        Kt[idx] = (__bf16)(loc + scale * eps_w[(size_t)t * NWc + widx]);
    }
    if (idx < GAMMAc * Hh) {        // sampled biases (f32)
        const int t = idx / Hh, j = idx % Hh;
        const int widx = Hh * Hh + j;
        const float loc   = post_t[widx];
        const float rho   = post_t[NWc + widx];
        const float scale = 1e-5f + 0.1f * softplus_(rho);
        bvec[idx] = loc + scale * eps_w[(size_t)t * NWc + widx];
    }
}

// ============ WMMA helpers (contiguous k order: lane half h holds k = 16h..16h+15, A and B alike) ============
__device__ __forceinline__ v16bf load_a_frag(const __bf16* __restrict__ rowbase,
                                             int kk, int kA)
{
    AFrag a;
    const __bf16* p = rowbase + kk + kA;
    a.h[0] = *reinterpret_cast<const v8bf*>(p);
    a.h[1] = *reinterpret_cast<const v8bf*>(p + 8);
    return a.v;
}

// z[16][512] = xh[16][192] @ WU[192][512]; B-fragments live in registers (wB).
__device__ __forceinline__ void lstm_gemm(const __bf16 (*__restrict__ sXH)[KDIM],
                                          const v16bf* __restrict__ wB,  // 24 frags
                                          float (*__restrict__ sZ)[NZ],
                                          int wave, int lane)
{
    const int lrow = lane & 15;
    const int kA = (lane < 16) ? 0 : 16;
    v8f acc[4] = {};
#pragma unroll
    for (int kk = 0; kk < 6; ++kk) {
        const v16bf a = load_a_frag(&sXH[lrow][0], kk * 32, kA);
#pragma unroll
        for (int j = 0; j < 4; ++j)
            acc[j] = __builtin_amdgcn_wmma_f32_16x16x32_bf16(
                false, a, false, wB[kk * 4 + j], (short)0, acc[j], false, false);
    }
#pragma unroll
    for (int j = 0; j < 4; ++j) {
        const int n = wave * 64 + j * 16 + lrow;
#pragma unroll
        for (int r = 0; r < 8; ++r) {
            const int m = (lane < 16) ? r : (8 + r);
            sZ[m][n] = acc[j][r];
        }
    }
}

// d[16][128] = h[16][128] @ K[128][128] + b   (Kt streamed into LDS)
__device__ __forceinline__ void dense_gemm(const __bf16 (*__restrict__ sXH)[KDIM],
                                           const __bf16* __restrict__ KtL,   // LDS
                                           const float* __restrict__ bv,
                                           float (*__restrict__ sD)[Hh],
                                           int wave, int lane)
{
    const int lrow = lane & 15;
    const int kA = (lane < 16) ? 0 : 16;
    const int kB = (lane < 16) ? 0 : 16;
    const int n = wave * 16 + lrow;
    v8f acc = {};
#pragma unroll
    for (int kk = 0; kk < Hh; kk += 32) {
        const v16bf a = load_a_frag(&sXH[lrow][Ff], kk, kA);
        const v16bf b = *reinterpret_cast<const v16bf*>(KtL + (size_t)n * Hh + kk + kB);
        acc = __builtin_amdgcn_wmma_f32_16x16x32_bf16(
            false, a, false, b, (short)0, acc, false, false);
    }
#pragma unroll
    for (int r = 0; r < 8; ++r) {
        const int m = (lane < 16) ? r : (8 + r);
        sD[m][n] = acc[r] + bv[n];
    }
}

// ============ persistent recurrence kernel: one 16-row batch tile per block ============
__global__ void __launch_bounds__(256)
lstm_forecast_kernel(const float* __restrict__ inputs,   // [B][T][F]
                     const float* __restrict__ lstm_b,   // [512]
                     const float* __restrict__ eps_y,    // [27][B][F]
                     const __bf16* __restrict__ WtUt,    // [512][192]
                     const __bf16* __restrict__ KtG,     // [28][128][128]
                     const float* __restrict__ bvec,     // [28][128]
                     float* __restrict__ out)            // [2][B][28][F]
{
    __shared__ __align__(16) __bf16 sXH[16][KDIM];    // fused [x|h] bf16  (6 KB)
    __shared__ __align__(16) float  sZ[16][NZ];       // pre-activations  (32 KB)
    __shared__ __align__(16) float  sC[16][Hh];       // cell state        (8 KB)
    __shared__ __align__(16) float  sD[16][Hh];       // dense output      (8 KB)
    __shared__ __align__(16) __bf16 sKt[2][Hh * Hh];  // Kt double-buffer (64 KB)
    __shared__ __align__(16) float  sXf32[16 * Ff];   // prefetched x row  (4 KB)
    __shared__ __align__(16) float  sEps[16 * Ff];    // prefetched eps_y  (4 KB)

    const int tid  = threadIdx.x;
    const int wave = tid >> 5;
    const int lane = tid & 31;
    const int lrow = lane & 15;
    const int b0   = blockIdx.x * 16;

    // ---- hoist LSTM weight B-fragments into registers (clobber-proof) ----
    v16bf wB[24];
    {
        const int kB = (lane < 16) ? 0 : 16;
#pragma unroll
        for (int kk = 0; kk < 6; ++kk)
#pragma unroll
            for (int j = 0; j < 4; ++j) {
                const int n = wave * 64 + j * 16 + lrow;
                wB[kk * 4 + j] = *reinterpret_cast<const v16bf*>(
                    WtUt + (size_t)n * KDIM + kk * 32 + kB);
            }
    }
    // ---- hoist per-thread gate biases (col invariant: 256 = 2*128) ----
    const int bcol = tid & 127;
    const float gb_i = lstm_b[bcol];
    const float gb_f = lstm_b[bcol + 128];
    const float gb_g = lstm_b[bcol + 256];
    const float gb_o = lstm_b[bcol + 384];

    // init h = c = 0
    for (int i = tid; i < 16 * Hh; i += 256) {
        sC[i >> 7][i & 127] = 0.0f;
        sXH[i >> 7][Ff + (i & 127)] = (__bf16)0.0f;
    }
    // prologue prefetch: x(0) and Kt[0]
    prefetch_input_row(inputs, b0, 0, sXf32, tid);
    prefetch_kt(KtG, &sKt[0][0], tid);
    wait_async0();
    __syncthreads();

    auto gates = [&]() {
#pragma unroll
        for (int k = 0; k < 8; ++k) {
            const int r = (tid >> 7) + 2 * k;           // 0..15
            const float zi = sZ[r][bcol]       + gb_i;
            const float zf = sZ[r][bcol + 128] + gb_f;
            const float zg = sZ[r][bcol + 256] + gb_g;
            const float zo = sZ[r][bcol + 384] + gb_o;
            const float c  = sig_(zf) * sC[r][bcol] + sig_(zi) * tanhf(zg);
            sC[r][bcol] = c;
            sXH[r][Ff + bcol] = (__bf16)(sig_(zo) * tanhf(c));
        }
    };
    auto write_out = [&](int g) {
        for (int i = tid; i < 16 * Ff; i += 256) {
            const int r = i >> 6, f = i & 63;
            const float loc = sD[r][f];
            const float sc  = 0.1f * softplus_(sD[r][Ff + f]);
            const size_t bb = (size_t)(b0 + r);
            out[(bb * GAMMAc + g) * Ff + f] = loc;
            out[(((size_t)Bsz + bb) * GAMMAc + g) * Ff + f] = sc;
        }
    };

    // -------- warm-up: 56 LSTM steps, input rows streamed asynchronously --------
    for (int t = 0; t < WARM; ++t) {
        for (int i = tid; i < 16 * Ff; i += 256)            // f32 -> bf16
            sXH[i >> 6][i & 63] = (__bf16)sXf32[i];
        __syncthreads();
        // overlap next-step streams with this step's WMMA work.
        prefetch_input_row(inputs, b0, t + 1, sXf32, tid);  // t=55 -> teacher row 56
        if (t == WARM - 1)
            prefetch_flat4k(eps_y + (size_t)b0 * Ff, sEps, tid);   // eps_y[0]
        lstm_gemm(sXH, wB, sZ, wave, lane);
        __syncthreads();
        gates();
        wait_async0();
        __syncthreads();
    }

    // -------- d0 + output g=0 (Kt[1] streams in behind the dense GEMM) --------
    prefetch_kt(KtG + (size_t)Hh * Hh, &sKt[1][0], tid);
    dense_gemm(sXH, &sKt[0][0], bvec, sD, wave, lane);
    __syncthreads();
    write_out(0);
    __syncthreads();

    // -------- 27 feedback steps --------
    for (int s = 0; s < GAMMAc - 1; ++s) {
        const int g = s + 1;
        // build next input x from d, prefetched eps_y[s] and teacher row
        for (int i = tid; i < 16 * Ff; i += 256) {
            const int r = i >> 6, f = i & 63;
            const float loc = sD[r][f];
            const float sc  = 0.1f * softplus_(sD[r][Ff + f]);
            const float y   = loc + sc * sEps[i];
            const float x   = (s < LAGc && f < Ff - 1) ? sXf32[i] : y;
            sXH[r][f] = (__bf16)x;
        }
        __syncthreads();
        // issue next-step streams (overlap with lstm_gemm + gates)
        if (s + 1 < GAMMAc - 1)
            prefetch_flat4k(eps_y + ((size_t)(s + 1) * Bsz + b0) * Ff, sEps, tid);
        if (s + 1 < LAGc)
            prefetch_input_row(inputs, b0, Tt - LAGc + s + 1, sXf32, tid);
        if (g + 1 < GAMMAc)
            prefetch_kt(KtG + (size_t)(g + 1) * Hh * Hh, &sKt[(g + 1) & 1][0], tid);
        lstm_gemm(sXH, wB, sZ, wave, lane);
        __syncthreads();
        gates();
        wait_async0();          // Kt[g] (issued last step) + this step's streams done
        __syncthreads();
        dense_gemm(sXH, &sKt[g & 1][0], bvec + g * Hh, sD, wave, lane);
        __syncthreads();
        write_out(g);
        __syncthreads();
    }
}

// ============ host entry ============
extern "C" void kernel_launch(void* const* d_in, const int* in_sizes, int n_in,
                              void* d_out, int out_size, void* d_ws, size_t ws_size,
                              hipStream_t stream)
{
    (void)in_sizes; (void)n_in; (void)out_size; (void)ws_size;
    const float* inputs = (const float*)d_in[0];
    const float* lstm_W = (const float*)d_in[1];
    const float* lstm_U = (const float*)d_in[2];
    const float* lstm_b = (const float*)d_in[3];
    const float* post_t = (const float*)d_in[4];
    const float* eps_w  = (const float*)d_in[5];
    const float* eps_y  = (const float*)d_in[6];

    __bf16* WtUt = (__bf16*)d_ws;
    __bf16* Kt   = (__bf16*)((char*)d_ws + WS_OFF_KT);
    float*  bvec = (float*)((char*)d_ws + WS_OFF_BVEC);
    float*  out  = (float*)d_out;

    precompute_kernel<<<(GAMMAc * Hh * Hh + 255) / 256, 256, 0, stream>>>(
        lstm_W, lstm_U, post_t, eps_w, WtUt, Kt, bvec);

    lstm_forecast_kernel<<<Bsz / 16, 256, 0, stream>>>(
        inputs, lstm_b, eps_y, WtUt, Kt, bvec, out);
}
